// Head_88940182766104
// MI455X (gfx1250) — matrix operand order repaired and checked by reading, not compiled, not run on hardware
//
#include <hip/hip_runtime.h>
#include <hip/hip_bf16.h>

#define BATCH 8
#define SEQ   4096
#define NE    1024
#define HALF_ 32
#define HS    64

typedef __attribute__((ext_vector_type(16))) __bf16 v16bf;
typedef __attribute__((ext_vector_type(8)))  __bf16 v8bf;
typedef __attribute__((ext_vector_type(8)))  float  v8f;

// workspace offsets in bf16 elements
#define Q_OFF   0ull
#define K_OFF   2097152ull   // 8*4096*64
#define V_OFF   4194304ull
#define W1T_OFF 6291456ull   // + 3*32*1024
#define W2T_OFF 6389760ull   // + 3*64*32  -> total 6395904 elems (~12.2 MB)

__device__ __forceinline__ v8f wmma_bf16(v16bf a, v16bf b, v8f c) {
  // D = A(16x32 bf16) * B(32x16 bf16) + C(16x16 f32)
  return __builtin_amdgcn_wmma_f32_16x16x32_bf16(false, a, false, b, (short)0, c, false, false);
}

// A-matrix 16x32 bf16 fragment (contiguous k order, matching B fills):
// lanes 0-15: row=lane, elems 0-15 = K 0..15
// lanes 16-31: row=lane-16, elems 0-15 = K 16..31
__device__ __forceinline__ v16bf load_a_frag(const __bf16* row, int kbase, bool hi) {
  const int o0 = kbase + (hi ? 16 : 0);
  const int o1 = kbase + (hi ? 24 : 8);
  v8bf lo = *(const v8bf*)(row + o0);
  v8bf h2 = *(const v8bf*)(row + o1);
  v16bf r;
#pragma unroll
  for (int i = 0; i < 8; ++i) { r[i] = lo[i]; r[8 + i] = h2[i]; }
  return r;
}

// ---------------------------------------------------------------------------
// Kernel 1: convert + transpose weights to bf16.
// w1T[m][n][k]  (m: 0=q,1=k,2=v)  n in [0,32), k in [0,1024)
// w2T[m][h][k]  h in [0,64), k in [0,32); q-branch folds scale*log2(e)
// ---------------------------------------------------------------------------
__global__ void setup_weights_kernel(const float* __restrict__ wk1, const float* __restrict__ wk2,
                                     const float* __restrict__ wq1, const float* __restrict__ wq2,
                                     const float* __restrict__ wv1, const float* __restrict__ wv2,
                                     __bf16* __restrict__ w1T, __bf16* __restrict__ w2T) {
  const int idx = blockIdx.x * blockDim.x + threadIdx.x;
  const int n1 = 3 * HALF_ * NE;   // 98304
  const int n2 = 3 * HS * HALF_;   // 6144
  if (idx < n1) {
    const int m   = idx / (HALF_ * NE);
    const int rem = idx % (HALF_ * NE);
    const int n   = rem / NE;
    const int k   = rem % NE;
    const float* src = (m == 0) ? wq1 : (m == 1) ? wk1 : wv1;
    w1T[idx] = (__bf16)src[k * HALF_ + n];
  } else if (idx < n1 + n2) {
    const int j   = idx - n1;
    const int m   = j / (HS * HALF_);
    const int rem = j % (HS * HALF_);
    const int h   = rem / HALF_;
    const int k   = rem % HALF_;
    const float* src = (m == 0) ? wq2 : (m == 1) ? wk2 : wv2;
    float f = src[k * HS + h];
    if (m == 0) f *= 0.125f * 1.44269504088896340736f; // 1/sqrt(64) * log2(e)
    w2T[j] = (__bf16)f;
  }
}

// ---------------------------------------------------------------------------
// Kernel 2: fused q/k/v projection.  One wave per 16-row tile of x.
//   h = tanh(x@W1 + b)  (16x32, WMMA over K=1024)
//   y = h @ W2          (16x64, via LDS transpose of h into A-layout)
// q,k stored row-major bf16 [B*S,64]; v stored transposed [B,64,S].
// ---------------------------------------------------------------------------
__global__ void qkv_proj_kernel(const float* __restrict__ x,
                                const float* __restrict__ bq,
                                const float* __restrict__ bk,
                                const float* __restrict__ bv,
                                const __bf16* __restrict__ w1T,
                                const __bf16* __restrict__ w2T,
                                __bf16* __restrict__ qb,
                                __bf16* __restrict__ kb,
                                __bf16* __restrict__ vT) {
  __shared__ __align__(32) __bf16 tbuf[4][16 * 32];
  const int  lane = threadIdx.x & 31;
  const int  wave = threadIdx.x >> 5;
  const bool hi   = lane >= 16;
  const int  n    = lane & 15;
  const int  off  = hi ? 8 : 0;
  const int  tile = blockIdx.x * 4 + wave;
  const int  row0 = tile * 16;
  const float* xrow = x + (size_t)(row0 + n) * NE;

  v8f acc[3][2];
#pragma unroll
  for (int m = 0; m < 3; ++m)
#pragma unroll
    for (int c = 0; c < 2; ++c)
      acc[m][c] = (v8f){};

  for (int kk = 0; kk < 32; ++kk) {
    const int o0 = kk * 32 + (hi ? 16 : 0);
    const int o1 = kk * 32 + (hi ? 24 : 8);
    float4 f0 = *(const float4*)(xrow + o0);
    float4 f1 = *(const float4*)(xrow + o0 + 4);
    float4 f2 = *(const float4*)(xrow + o1);
    float4 f3 = *(const float4*)(xrow + o1 + 4);
    v16bf a;
    a[0]=(__bf16)f0.x; a[1]=(__bf16)f0.y; a[2]=(__bf16)f0.z; a[3]=(__bf16)f0.w;
    a[4]=(__bf16)f1.x; a[5]=(__bf16)f1.y; a[6]=(__bf16)f1.z; a[7]=(__bf16)f1.w;
    a[8]=(__bf16)f2.x; a[9]=(__bf16)f2.y; a[10]=(__bf16)f2.z; a[11]=(__bf16)f2.w;
    a[12]=(__bf16)f3.x; a[13]=(__bf16)f3.y; a[14]=(__bf16)f3.z; a[15]=(__bf16)f3.w;
#pragma unroll
    for (int m = 0; m < 3; ++m) {
#pragma unroll
      for (int c = 0; c < 2; ++c) {
        const __bf16* bp = w1T + (size_t)m * (HALF_ * NE)
                               + (size_t)(c * 16 + n) * NE
                               + kk * 32 + (hi ? 16 : 0);
        acc[m][c] = wmma_bf16(a, *(const v16bf*)bp, acc[m][c]);
      }
    }
  }

  const float* biases[3] = {bq, bk, bv};
#pragma unroll
  for (int m = 0; m < 3; ++m) {
    // bias + tanh, write C-layout tile transposed into LDS [row][col]
#pragma unroll
    for (int c = 0; c < 2; ++c) {
      const float bcol = biases[m][c * 16 + n];
#pragma unroll
      for (int j = 0; j < 8; ++j) {
        float t = tanhf(acc[m][c][j] + bcol);
        tbuf[wave][(j + off) * 32 + c * 16 + n] = (__bf16)t;
      }
    }
    // reload as A-layout 16x32
    v16bf a2 = load_a_frag(&tbuf[wave][n * 32], 0, hi);
#pragma unroll
    for (int c2 = 0; c2 < 4; ++c2) {
      const __bf16* bp = w2T + (size_t)m * (HS * HALF_)
                             + (size_t)(c2 * 16 + n) * HALF_
                             + (hi ? 16 : 0);
      v8f y = (v8f){};
      y = wmma_bf16(a2, *(const v16bf*)bp, y);
#pragma unroll
      for (int j = 0; j < 8; ++j) {
        const int rr  = row0 + j + off;
        const int col = c2 * 16 + n;
        if (m == 0) {
          qb[(size_t)rr * HS + col] = (__bf16)y[j];
        } else if (m == 1) {
          kb[(size_t)rr * HS + col] = (__bf16)y[j];
        } else {
          const int bi = rr >> 12;          // SEQ = 4096
          const int s  = rr & (SEQ - 1);
          vT[((size_t)bi * HS + col) * SEQ + s] = (__bf16)y[j];
        }
      }
    }
  }
}

// ---------------------------------------------------------------------------
// Kernel 3: causal flash attention.  grid = (SEQ/64, BATCH), 4 waves/WG,
// each wave owns 16 query rows and streams 32-key steps up to the diagonal.
// Scores are already in exp2 domain (scale*log2e folded into Wq2).
// ---------------------------------------------------------------------------
__global__ void attn_kernel(const __bf16* __restrict__ qb,
                            const __bf16* __restrict__ kb,
                            const __bf16* __restrict__ vT,
                            float* __restrict__ out) {
  __shared__ __align__(32) __bf16 pbuf[4][16 * 32];
  const int  lane = threadIdx.x & 31;
  const int  wave = threadIdx.x >> 5;
  const bool hi   = lane >= 16;
  const int  n    = lane & 15;
  const int  off  = hi ? 8 : 0;
  const int  b    = blockIdx.y;
  const int  row0 = blockIdx.x * 64 + wave * 16;

  const __bf16* qrow = qb + ((size_t)b * SEQ + row0 + n) * HS;
  v16bf qf0 = load_a_frag(qrow, 0, hi);
  v16bf qf1 = load_a_frag(qrow, 32, hi);

  v8f o0 = (v8f){}, o1 = (v8f){}, o2 = (v8f){}, o3 = (v8f){};
  float m_run[8], l_run[8];
#pragma unroll
  for (int j = 0; j < 8; ++j) { m_run[j] = -1e30f; l_run[j] = 0.0f; }

  const __bf16* kbase = kb + (size_t)b * SEQ * HS;
  const __bf16* vbase = vT + (size_t)b * HS * SEQ;

  const int nsteps = (row0 + 47) >> 5;   // keys up to row0+15 inclusive
  for (int kbi = 0; kbi < nsteps; ++kbi) {
    const int key0 = kbi * 32;

    // K^T B-fragments: lane=key column, 16 contiguous head elems per lane
    const __bf16* kr0 = kbase + (size_t)(key0 + n) * HS + (hi ? 16 : 0);
    const __bf16* kr1 = kbase + (size_t)(key0 + 16 + n) * HS + (hi ? 16 : 0);
    v16bf kf00 = *(const v16bf*)(kr0);
    v16bf kf01 = *(const v16bf*)(kr0 + 32);
    v16bf kf10 = *(const v16bf*)(kr1);
    v16bf kf11 = *(const v16bf*)(kr1 + 32);

    if (kbi + 1 < nsteps) {
      __builtin_prefetch(kbase + (size_t)(key0 + 32 + n) * HS, 0, 1);
      __builtin_prefetch(vbase + (size_t)n * SEQ + key0 + 32, 0, 1);
    }

    v8f s0 = (v8f){}, s1 = (v8f){};
    s0 = wmma_bf16(qf0, kf00, s0);
    s0 = wmma_bf16(qf1, kf01, s0);
    s1 = wmma_bf16(qf0, kf10, s1);
    s1 = wmma_bf16(qf1, kf11, s1);

    // online softmax per row (rows live in vector elements; keys across lanes)
#pragma unroll
    for (int j = 0; j < 8; ++j) {
      const int row = row0 + j + off;
      float a0 = (key0 + n      <= row) ? s0[j] : -1e30f;
      float a1 = (key0 + 16 + n <= row) ? s1[j] : -1e30f;
      float mx = fmaxf(a0, a1);
      mx = fmaxf(mx, __shfl_xor(mx, 1, 16));
      mx = fmaxf(mx, __shfl_xor(mx, 2, 16));
      mx = fmaxf(mx, __shfl_xor(mx, 4, 16));
      mx = fmaxf(mx, __shfl_xor(mx, 8, 16));
      const float mnew  = fmaxf(m_run[j], mx);
      const float alpha = exp2f(m_run[j] - mnew);
      const float p0 = exp2f(a0 - mnew);
      const float p1 = exp2f(a1 - mnew);
      float rs = p0 + p1;
      rs += __shfl_xor(rs, 1, 16);
      rs += __shfl_xor(rs, 2, 16);
      rs += __shfl_xor(rs, 4, 16);
      rs += __shfl_xor(rs, 8, 16);
      l_run[j] = l_run[j] * alpha + rs;
      m_run[j] = mnew;
      o0[j] *= alpha; o1[j] *= alpha; o2[j] *= alpha; o3[j] *= alpha;
      // stage P (bf16) for the LDS transpose into A-layout
      pbuf[wave][(j + off) * 32 + n]      = (__bf16)p0;
      pbuf[wave][(j + off) * 32 + 16 + n] = (__bf16)p1;
    }

    // P as A-matrix 16x32 (DS ops are in-order within a wave)
    v16bf pf = load_a_frag(&pbuf[wave][n * 32], 0, hi);

    // V B-fragments from transposed V: lane=head column, contiguous keys
    const __bf16* vr = vbase + (size_t)n * SEQ + key0 + (hi ? 16 : 0);
    o0 = wmma_bf16(pf, *(const v16bf*)(vr),            o0);
    o1 = wmma_bf16(pf, *(const v16bf*)(vr + 16 * SEQ), o1);
    o2 = wmma_bf16(pf, *(const v16bf*)(vr + 32 * SEQ), o2);
    o3 = wmma_bf16(pf, *(const v16bf*)(vr + 48 * SEQ), o3);
  }

  float* orow = out + ((size_t)b * SEQ + row0) * HS;
#pragma unroll
  for (int j = 0; j < 8; ++j) {
    const float inv = 1.0f / l_run[j];
    float* p = orow + (size_t)(j + off) * HS + n;
    p[0]  = o0[j] * inv;
    p[16] = o1[j] * inv;
    p[32] = o2[j] * inv;
    p[48] = o3[j] * inv;
  }
}

// ---------------------------------------------------------------------------
extern "C" void kernel_launch(void* const* d_in, const int* in_sizes, int n_in,
                              void* d_out, int out_size, void* d_ws, size_t ws_size,
                              hipStream_t stream) {
  (void)in_sizes; (void)n_in; (void)out_size; (void)ws_size;
  const float* x   = (const float*)d_in[0];
  const float* wk1 = (const float*)d_in[1];
  const float* bk1 = (const float*)d_in[2];
  const float* wk2 = (const float*)d_in[3];
  const float* wq1 = (const float*)d_in[4];
  const float* bq1 = (const float*)d_in[5];
  const float* wq2 = (const float*)d_in[6];
  const float* wv1 = (const float*)d_in[7];
  const float* bv1 = (const float*)d_in[8];
  const float* wv2 = (const float*)d_in[9];
  float* out = (float*)d_out;

  __bf16* ws  = (__bf16*)d_ws;
  __bf16* qb  = ws + Q_OFF;
  __bf16* kb  = ws + K_OFF;
  __bf16* vT  = ws + V_OFF;
  __bf16* w1T = ws + W1T_OFF;
  __bf16* w2T = ws + W2T_OFF;

  // 1) weights -> bf16, transposed for B-fragment loads (scale folded into Wq2)
  setup_weights_kernel<<<408, 256, 0, stream>>>(wk1, wk2, wq1, wq2, wv1, wv2, w1T, w2T);

  // 2) fused q/k/v projection: 2048 row-tiles, 4 waves per 128-thread block
  qkv_proj_kernel<<<(BATCH * SEQ / 16) / 4, 128, 0, stream>>>(
      x, bq1, bk1, bv1, w1T, w2T, qb, kb, vT);

  // 3) causal flash attention
  attn_kernel<<<dim3(SEQ / 64, BATCH), 128, 0, stream>>>(qb, kb, vT, out);
}
